// GuideMatrixGenerator_5248450035890
// MI455X (gfx1250) — matrix operand order repaired and checked by reading, not compiled, not run on hardware
//
#include <hip/hip_runtime.h>
#include <math.h>

#define B_   512
#define C_   256
#define L_   196
#define D_   512
#define LPAD 224               // padded spatial (rows of samn_T / attn, K of GEMM2)
#define NPAD 208               // padded N (cols of attn)
#define CL_  (C_ * L_)         // 50176

typedef float  v8f   __attribute__((ext_vector_type(8)));
typedef __bf16 v8bf  __attribute__((ext_vector_type(8)));
typedef __bf16 v16bf __attribute__((ext_vector_type(16)));
typedef unsigned int u32x4 __attribute__((ext_vector_type(4)));
typedef unsigned int u32x8 __attribute__((ext_vector_type(8)));

union V16U { v16bf v; v8bf h[2]; };

__device__ __forceinline__ unsigned short f2bf(float f) {
  unsigned int u = __float_as_uint(f);
  u += 0x7FFFu + ((u >> 16) & 1u);           // round-to-nearest-even
  return (unsigned short)(u >> 16);
}

// Build a TDM D# (groups 0/1) for a 1-row 2D copy of `nelem` 4-byte elements.
__device__ __forceinline__ void tdm_build_desc(unsigned long long gaddr,
                                               unsigned int ldsaddr,
                                               unsigned int nelem,
                                               u32x4* g0, u32x8* g1) {
  u32x4 a;
  a.x = 1u;                                               // count=1, user mode
  a.y = ldsaddr;                                          // lds_addr (bytes)
  a.z = (unsigned int)gaddr;                              // global_addr[31:0]
  a.w = (unsigned int)((gaddr >> 32) & 0x1FFFFFFull)      // global_addr[56:32]
        | (2u << 30);                                     // type=2 ("image")
  u32x8 b;
  b[0] = (2u << 16);                                      // data_size=2 -> 4 bytes
  b[1] = (nelem & 0xFFFFu) << 16;                         // tensor_dim0 lo16
  b[2] = ((nelem >> 16) & 0xFFFFu) | (1u << 16);          // dim0 hi16 | tensor_dim1=1 lo16
  b[3] = (nelem & 0xFFFFu) << 16;                         // dim1 hi16=0 | tile_dim0
  b[4] = 1u;                                              // tile_dim1=1, tile_dim2=0
  b[5] = nelem;                                           // tensor_dim0_stride lo32
  b[6] = (nelem & 0xFFFFu) << 16;                         // stride0 hi16=0 | dim1_stride lo16
  b[7] = 0u;                                              // dim1_stride hi32
  *g0 = a;
  *g1 = b;
}

// ---------------------------------------------------------------------------
// Kernel 1: similarity weights  simw[b][l] = cross(b) * sigmoid(imf_n.Wsim[l]+b[l]) / 10
// ---------------------------------------------------------------------------
__global__ void __launch_bounds__(256) simw_kernel(
    const float* __restrict__ imf, const float* __restrict__ txt,
    const float* __restrict__ Wsim, const float* __restrict__ bsim,
    float* __restrict__ simw) {
  __shared__ float ims[D_];
  __shared__ float sI[256], sT[256], sD[256];
  const int tid = threadIdx.x;
  const int b   = blockIdx.x;

  float aI = 0.f, aT = 0.f, aD = 0.f;
  for (int i = tid; i < D_; i += 256) {
    float v = imf[(size_t)b * D_ + i];
    float t = txt[(size_t)b * D_ + i];
    ims[i] = v;
    aI += v * v; aT += t * t; aD += v * t;
  }
  sI[tid] = aI; sT[tid] = aT; sD[tid] = aD;
  __syncthreads();
  for (int s = 128; s > 0; s >>= 1) {
    if (tid < s) { sI[tid] += sI[tid + s]; sT[tid] += sT[tid + s]; sD[tid] += sD[tid + s]; }
    __syncthreads();
  }
  const float ni    = fmaxf(sqrtf(sI[0]), 1e-12f);
  const float ntn   = fmaxf(sqrtf(sT[0]), 1e-12f);
  const float cross = sD[0] / (ni * ntn);

  for (int l = tid; l < L_; l += 256) {
    const float* w = Wsim + (size_t)l * D_;
    float d = 0.f;
    #pragma unroll 4
    for (int i = 0; i < D_; ++i) d += ims[i] * w[i];
    float z  = d / ni + bsim[l];
    float sg = 1.f / (1.f + __expf(-z));
    simw[(size_t)b * L_ + l] = cross * sg * 0.1f;
  }
}

// ---------------------------------------------------------------------------
// Kernel 2: spatial self-attention via bf16 WMMA, fused combine + sim scale.
// LDS layout (bytes):
//   snT : bf16[224][256]  @ 0        (GEMM1 A operand: samn^T, lane=M layout)
//   snC : bf16[256][224]  @ 114688   (GEMM2 A operand; GEMM1 B uses snT)
//   atn : bf16[224][208]  @ 229376   (GEMM2 B operand, norm folded in)
//   nrm : f32 [224]       @ 322560
//   swv : f32 [208]       @ 323456
// total = 324288 B (<= 320KB WGP LDS)
// ---------------------------------------------------------------------------
#define SNT_OFF   0
#define SNC_OFF   114688
#define ATN_OFF   229376
#define NRM_OFF   322560
#define SWV_OFF   323456
#define K2_LDS    324288

__global__ void __launch_bounds__(256) attn_kernel(
    const float* __restrict__ sam, const float* __restrict__ simw,
    float* __restrict__ out) {
  extern __shared__ char smem[];
  unsigned short* snT = (unsigned short*)(smem + SNT_OFF);
  unsigned short* snC = (unsigned short*)(smem + SNC_OFF);
  unsigned short* atn = (unsigned short*)(smem + ATN_OFF);
  float*          nrm = (float*)(smem + NRM_OFF);
  float*          swv = (float*)(smem + SWV_OFF);

  const int tid  = threadIdx.x;
  const int wave = tid >> 5;
  const int lane = tid & 31;
  const int b    = blockIdx.x;
  const float* samb = sam + (size_t)b * CL_;

  // ---- init / zero padding ----
  for (int i = tid; i < LPAD; i += 256) nrm[i] = 0.f;
  for (int i = tid; i < NPAD; i += 256) swv[i] = (i < L_) ? simw[(size_t)b * L_ + i] : 0.f;
  for (int i = tid; i < (LPAD - L_) * C_; i += 256) snT[L_ * C_ + i] = 0;
  for (int i = tid; i < C_ * (LPAD - L_); i += 256) {
    int c = i / (LPAD - L_), l = L_ + i % (LPAD - L_);
    snC[c * LPAD + l] = 0;
  }
  __syncthreads();

  // ---- pass 1: per-spatial-column sum of squares over channels ----
  {
    float part[7] = {0.f, 0.f, 0.f, 0.f, 0.f, 0.f, 0.f};
    for (int r = 0; r < 32; ++r) {
      const float* row = samb + (size_t)(wave * 32 + r) * L_;
      #pragma unroll
      for (int k = 0; k < 6; ++k) { float v = row[lane + 32 * k]; part[k] += v * v; }
      if (lane < 4) { float v = row[192 + lane]; part[6] += v * v; }
    }
    #pragma unroll
    for (int k = 0; k < 6; ++k) atomicAdd(&nrm[lane + 32 * k], part[k]);
    if (lane < 4) atomicAdd(&nrm[192 + lane], part[6]);
  }
  __syncthreads();
  for (int i = tid; i < LPAD; i += 256) nrm[i] = fmaxf(sqrtf(nrm[i]), 1e-12f);
  __syncthreads();

  // ---- pass 2: write L2-normalized bf16 in both layouts ----
  {
    float inv[7];
    #pragma unroll
    for (int k = 0; k < 7; ++k) {
      int l = lane + 32 * k;
      inv[k] = (l < L_) ? 1.f / nrm[l] : 0.f;
    }
    for (int r = 0; r < 32; ++r) {
      int c = wave * 32 + r;
      const float* row = samb + (size_t)c * L_;
      #pragma unroll
      for (int k = 0; k < 7; ++k) {
        int l = lane + 32 * k;
        if (l < L_) {
          unsigned short h = f2bf(row[l] * inv[k]);
          snC[c * LPAD + l] = h;
          snT[l * C_ + c]   = h;
        }
      }
    }
  }
  __syncthreads();

  // ---- GEMM1 (scores = samn^T . samn / 16) + softmax + fold value-norm ----
  #pragma unroll 1
  for (int mt = wave; mt < 14; mt += 8) {
    const int m0 = mt * 16;
    v8f zero = {0.f, 0.f, 0.f, 0.f, 0.f, 0.f, 0.f, 0.f};
    v8f acc[13];
    #pragma unroll
    for (int nt = 0; nt < 13; ++nt) acc[nt] = zero;

    const int arow = (m0 + (lane & 15)) * C_;
    const int asub = ((lane >> 4) & 1) * 8;
    #pragma unroll 1
    for (int kt = 0; kt < 8; ++kt) {
      V16U a;
      a.h[0] = *(const v8bf*)(snT + arow + kt * 32 + asub);
      a.h[1] = *(const v8bf*)(snT + arow + kt * 32 + asub + 16);
      const int brow = (lane & 15) * C_ + kt * 32 + asub;
      #pragma unroll
      for (int nt = 0; nt < 13; ++nt) {
        V16U bv;
        bv.h[0] = *(const v8bf*)(snT + brow + nt * 16 * C_);
        bv.h[1] = *(const v8bf*)(snT + brow + nt * 16 * C_ + 16);
        acc[nt] = __builtin_amdgcn_wmma_f32_16x16x32_bf16(
            false, a.v, false, bv.v, (short)0, acc[nt], false, false);
      }
    }

    // scale by 1/sqrt(C), mask padded columns (m in [196,208))
    #pragma unroll
    for (int nt = 0; nt < 13; ++nt)
      #pragma unroll
      for (int j = 0; j < 8; ++j) acc[nt][j] *= 0.0625f;
    if ((lane & 15) >= 4) {
      #pragma unroll
      for (int j = 0; j < 8; ++j) acc[12][j] = -1e30f;
    }

    // row softmax in C/D register layout (16-lane butterfly reductions)
    float rmax[8], rsum[8];
    #pragma unroll
    for (int j = 0; j < 8; ++j) {
      float m = acc[0][j];
      #pragma unroll
      for (int nt = 1; nt < 13; ++nt) m = fmaxf(m, acc[nt][j]);
      #pragma unroll
      for (int off = 1; off <= 8; off <<= 1) m = fmaxf(m, __shfl_xor(m, off, 32));
      rmax[j] = m;
    }
    #pragma unroll
    for (int j = 0; j < 8; ++j) {
      float s = 0.f;
      #pragma unroll
      for (int nt = 0; nt < 13; ++nt) {
        float e = __expf(acc[nt][j] - rmax[j]);
        acc[nt][j] = e;
        s += e;
      }
      #pragma unroll
      for (int off = 1; off <= 8; off <<= 1) s += __shfl_xor(s, off, 32);
      rsum[j] = s;
    }
    const int lhalf = (lane >> 4) & 1;
    #pragma unroll
    for (int j = 0; j < 8; ++j) {
      const int l = m0 + j + 8 * lhalf;
      const float sca = nrm[l] / rsum[j];            // fold value norm: sam = samn * nrm
      unsigned short* dst = atn + (lane & 15) * LPAD + l;
      #pragma unroll
      for (int nt = 0; nt < 13; ++nt) dst[nt * 16 * LPAD] = f2bf(acc[nt][j] * sca);
    }
  }
  __syncthreads();

  // ---- GEMM2 (weighted = samn . attn') + fused combine/scale epilogue ----
  const size_t outb = (size_t)b * CL_;
  #pragma unroll 1
  for (int t = wave; t < 16 * 13; t += 8) {
    const int ct = t / 13, nt = t % 13;
    const int c0 = ct * 16, n0 = nt * 16;
    v8f acc = {0.f, 0.f, 0.f, 0.f, 0.f, 0.f, 0.f, 0.f};
    const int arow = (c0 + (lane & 15)) * LPAD;
    const int asub = ((lane >> 4) & 1) * 8;
    #pragma unroll
    for (int kt = 0; kt < 7; ++kt) {
      V16U a, bv;
      a.h[0] = *(const v8bf*)(snC + arow + kt * 32 + asub);
      a.h[1] = *(const v8bf*)(snC + arow + kt * 32 + asub + 16);
      const int brow = (n0 + (lane & 15)) * LPAD + kt * 32 + asub;
      bv.h[0] = *(const v8bf*)(atn + brow);
      bv.h[1] = *(const v8bf*)(atn + brow + 16);
      acc = __builtin_amdgcn_wmma_f32_16x16x32_bf16(
          false, a.v, false, bv.v, (short)0, acc, false, false);
    }
    const int m = n0 + (lane & 15);
    if (m < L_) {
      const float sw = swv[m] * 0.5f;
      const int lh8 = ((lane >> 4) & 1) * 8;
      #pragma unroll
      for (int j = 0; j < 8; ++j) {
        const int c = c0 + j + lh8;
        const float s = samb[c * L_ + m];
        out[outb + c * L_ + m] = (s + acc[j]) * sw;
      }
    }
  }
}

// ---------------------------------------------------------------------------
// Kernel 3: hierarchical norm (channel LN, then spatial standardization, ddof=1)
// Whole (C,L) slab resident in LDS; moved with the Tensor Data Mover:
//   tensor_load_to_lds  : global guide slab -> LDS       (TENSORcnt)
//   tensor_store_from_lds: normalized slab  -> global    (TENSORcnt)
// ---------------------------------------------------------------------------
#define K3_LDS (CL_ * 4 + L_ * 4 + L_ * 4 + C_ * 4 + C_ * 4)   // 204320 B

__global__ void __launch_bounds__(256) hnorm_kernel(float* __restrict__ out) {
  extern __shared__ char smem[];
  float* x  = (float*)smem;
  float* mu = (float*)(smem + CL_ * 4);
  float* iv = mu + L_;
  float* m2 = iv + L_;
  float* d2 = m2 + C_;
  const int tid = threadIdx.x;
  const int b   = blockIdx.x;
  float* ob = out + (size_t)b * CL_;

  // LDS byte offset of the slab (addrspace(3) pointers are LDS offsets)
  const unsigned int ldsoff =
      (unsigned int)(size_t)((__attribute__((address_space(3))) char*)smem);

  // ---- TDM: DMA the whole slab global -> LDS (issued by wave 0 only) ----
  if (tid < 32) {
    u32x4 g0; u32x8 g1;
    tdm_build_desc((unsigned long long)(size_t)ob, ldsoff, (unsigned int)CL_, &g0, &g1);
    asm volatile("tensor_load_to_lds %0, %1"
                 :: "s"(g0), "s"(g1) : "memory");
    __builtin_amdgcn_s_wait_tensorcnt(0);
  }
  __syncthreads();

  // ---- stage 1: channel LayerNorm stats per spatial position ----
  if (tid < L_) {
    float s = 0.f, s2 = 0.f;
    for (int c = 0; c < C_; ++c) { float v = x[c * L_ + tid]; s += v; s2 += v * v; }
    float m   = s * (1.f / (float)C_);
    float var = s2 * (1.f / (float)C_) - m * m;
    mu[tid] = m;
    iv[tid] = rsqrtf(var + 1e-5f);
  }
  __syncthreads();

  // ---- stage 2: apply LN, spatial stats per channel (Bessel) ----
  {
    const int c = tid;
    float s = 0.f, s2 = 0.f;
    for (int l = 0; l < L_; ++l) {
      float v = (x[c * L_ + l] - mu[l]) * iv[l];
      x[c * L_ + l] = v;
      s += v; s2 += v * v;
    }
    float m   = s * (1.f / (float)L_);
    float var = (s2 - (float)L_ * m * m) * (1.f / (float)(L_ - 1));
    var = fmaxf(var, 0.f);
    m2[c] = m;
    d2[c] = 1.f / (sqrtf(var) + 1e-6f);
  }
  __syncthreads();

  // ---- stage 3: final standardization written back into LDS ----
  for (int i = tid; i < CL_; i += 256) {
    int c = i / L_;
    x[i] = (x[i] - m2[c]) * d2[c];
  }
  __syncthreads();

  // ---- TDM: DMA the normalized slab LDS -> global ----
  if (tid < 32) {
    u32x4 g0; u32x8 g1;
    tdm_build_desc((unsigned long long)(size_t)ob, ldsoff, (unsigned int)CL_, &g0, &g1);
    asm volatile("tensor_store_from_lds %0, %1"
                 :: "s"(g0), "s"(g1) : "memory");
    __builtin_amdgcn_s_wait_tensorcnt(0);
  }
}

// ---------------------------------------------------------------------------
extern "C" void kernel_launch(void* const* d_in, const int* in_sizes, int n_in,
                              void* d_out, int out_size, void* d_ws, size_t ws_size,
                              hipStream_t stream) {
  const float* imf  = (const float*)d_in[0];
  const float* txt  = (const float*)d_in[1];
  const float* sam  = (const float*)d_in[2];
  const float* Wsim = (const float*)d_in[3];
  const float* bsim = (const float*)d_in[4];
  float* out  = (float*)d_out;
  float* simw = (float*)d_ws;   // B_*L_ floats of scratch

  hipFuncSetAttribute((const void*)attn_kernel,
                      hipFuncAttributeMaxDynamicSharedMemorySize, K2_LDS);
  hipFuncSetAttribute((const void*)hnorm_kernel,
                      hipFuncAttributeMaxDynamicSharedMemorySize, K3_LDS);

  simw_kernel<<<B_, 256, 0, stream>>>(imf, txt, Wsim, bsim, simw);
  attn_kernel<<<B_, 256, K2_LDS, stream>>>(sam, simw, out);
  hnorm_kernel<<<B_, 256, K3_LDS, stream>>>(out);
}
